// PNP_704374637376
// MI455X (gfx1250) — matrix operand order repaired and checked by reading, not compiled, not run on hardware
//
#include <hip/hip_runtime.h>
#include <hip/hip_bf16.h>

// ---------------------------------------------------------------------------
// MI455X (gfx1250) PnP attention. B=3, S=4096, D=320, 8 heads x d=40.
// Attention = 64 GFLOP (dominant) -> flash attention on v_wmma_f32_16x16x32_f16
// (f32 accumulate); unique data ~50MB lives in 192MB L2 -> compute bound.
// Projections use the fp32 matrix pipe (v_wmma_f32_16x16x4_f32) for accuracy.
// d=40 zero-padded to 64 so f16 WMMA K dim is exactly 2x32.
// K tiles staged with GLOBAL_LOAD_ASYNC_TO_LDS_B128 (ASYNCcnt path).
// ---------------------------------------------------------------------------

#define USE_ASYNC_LDS 1

typedef __attribute__((ext_vector_type(16))) _Float16 v16h;
typedef __attribute__((ext_vector_type(8)))  _Float16 v8h;
typedef __attribute__((ext_vector_type(8)))  float    v8f;
typedef __attribute__((ext_vector_type(2)))  float    v2f;

#define N_HEADS 8
#define D_HEAD  40
#define D_PAD   64
#define SEQ     4096
#define D_MODEL 320
#define BATCH   3
#define QK_SCALE 0.15811388300841897f  // 40^-0.5
#define QTILES  2                      // query tiles (of 16 rows) per wave

static __device__ __forceinline__ v8f wmma_f16(v16h a, v16h b, v8f c) {
  return __builtin_amdgcn_wmma_f32_16x16x32_f16(false, a, false, b, (short)0, c,
                                                false, false);
}
static __device__ __forceinline__ v8f wmma_f32(v2f a, v2f b, v8f c) {
  return __builtin_amdgcn_wmma_f32_16x16x4_f32(false, a, false, b, (short)0, c,
                                               false, false);
}
static __device__ __forceinline__ v16h cat8(v8h lo, v8h hi) {
  return __builtin_shufflevector(lo, hi, 0, 1, 2, 3, 4, 5, 6, 7,
                                         8, 9, 10, 11, 12, 13, 14, 15);
}

// ---------------------------------------------------------------------------
// Y = (X[M,320] @ W^T) * scale -> f16 padded head planes [(b*8+h)][s][64].
// fp32 WMMA 16x16x4; one 16x16 tile per wave.
// ---------------------------------------------------------------------------
__global__ __launch_bounds__(256) void proj_heads_f16(
    const float* __restrict__ X, const float* __restrict__ W,
    _Float16* __restrict__ outH, float scale) {
  const int ntN = D_MODEL / 16;
  int wave = threadIdx.x >> 5, lane = threadIdx.x & 31;
  int hi = lane >> 4, lm = lane & 15;
  int tile = blockIdx.x * 8 + wave;
  int m0 = (tile / ntN) * 16, n0 = (tile % ntN) * 16;

  const float* xr = X + (size_t)(m0 + lm) * D_MODEL + hi * 2;
  const float* wr = W + (size_t)(n0 + lm) * D_MODEL + hi * 2;
  v8f c = {};
#pragma unroll 4
  for (int k0 = 0; k0 < D_MODEL; k0 += 4) {
    v2f a = *(const v2f*)(xr + k0);
    v2f b = *(const v2f*)(wr + k0);
    c = wmma_f32(a, b, c);
  }
  int e = n0 + lm, head = e / D_HEAD, cc = e % D_HEAD;
#pragma unroll
  for (int r = 0; r < 8; ++r) {
    int row = m0 + r + hi * 8;
    int bidx = row / SEQ, s = row % SEQ;
    size_t idx = ((size_t)(bidx * N_HEADS + head) * SEQ + s) * D_PAD + cc;
    outH[idx] = (_Float16)(c[r] * scale);
  }
}

// ---------------------------------------------------------------------------
// Y = Xo[12288,320] @ Wo^T + bo, fp32 WMMA, fp32 output.
// ---------------------------------------------------------------------------
__global__ __launch_bounds__(256) void proj_out_f32(
    const float* __restrict__ X, const float* __restrict__ W,
    const float* __restrict__ bias, float* __restrict__ Y) {
  const int ntN = D_MODEL / 16;
  int wave = threadIdx.x >> 5, lane = threadIdx.x & 31;
  int hi = lane >> 4, lm = lane & 15;
  int tile = blockIdx.x * 8 + wave;
  int m0 = (tile / ntN) * 16, n0 = (tile % ntN) * 16;

  const float* xr = X + (size_t)(m0 + lm) * D_MODEL + hi * 2;
  const float* wr = W + (size_t)(n0 + lm) * D_MODEL + hi * 2;
  v8f c = {};
#pragma unroll 4
  for (int k0 = 0; k0 < D_MODEL; k0 += 4) {
    v2f a = *(const v2f*)(xr + k0);
    v2f b = *(const v2f*)(wr + k0);
    c = wmma_f32(a, b, c);
  }
  int e = n0 + lm;
  float bv = bias[e];
#pragma unroll
  for (int r = 0; r < 8; ++r) {
    int row = m0 + r + hi * 8;
    Y[(size_t)row * D_MODEL + e] = c[r] + bv;
  }
}

// ---------------------------------------------------------------------------
// Flash attention. grid = (SEQ/(128*QTILES), 24), 256 threads = 8 waves.
// Each wave owns QTILES 16-row query tiles; the block shares 32-key K/V tiles
// staged in LDS (K via async-to-LDS DMA, V transposed through registers).
// ---------------------------------------------------------------------------
__global__ __launch_bounds__(256) void flash_attn(
    const _Float16* __restrict__ qh, const _Float16* __restrict__ kh,
    const _Float16* __restrict__ vh, float* __restrict__ oat) {
  __shared__ _Float16 Ksh[32 * D_PAD];          // [key][feature]
  __shared__ _Float16 Vtsh[D_PAD * 32];         // [feature][key]
  __shared__ _Float16 Psh[8][QTILES][16 * 32];  // per-wave P relayout scratch

  int bh = blockIdx.y;
  int b = bh >> 3, h = bh & 7;
  int wave = threadIdx.x >> 5, lane = threadIdx.x & 31;
  int hi = lane >> 4, lm = lane & 15;
  int t = threadIdx.x;
  int q0 = blockIdx.x * (128 * QTILES) + wave * (16 * QTILES);

  // PnP injection: Q/K come from the source batch (batch-0 planes).
  const _Float16* Q = qh + (size_t)h * SEQ * D_PAD;
  const _Float16* K = kh + (size_t)h * SEQ * D_PAD;
  const _Float16* V = vh + (size_t)bh * SEQ * D_PAD;

  // Q A-fragments (Q is pre-scaled by QK_SCALE at projection time).
  v16h qa[QTILES][2];
#pragma unroll
  for (int qt = 0; qt < QTILES; ++qt)
#pragma unroll
    for (int kc = 0; kc < 2; ++kc) {
      const _Float16* base =
          Q + (size_t)(q0 + qt * 16 + lm) * D_PAD + kc * 32 + hi * 8;
      qa[qt][kc] = cat8(*(const v8h*)base, *(const v8h*)(base + 16));
    }

  v8f zero = {};
  v8f acc[QTILES][4];
  float m_run[QTILES][8], l_run[QTILES][8];
#pragma unroll
  for (int qt = 0; qt < QTILES; ++qt) {
#pragma unroll
    for (int i = 0; i < 4; ++i) acc[qt][i] = zero;
#pragma unroll
    for (int r = 0; r < 8; ++r) { m_run[qt][r] = -INFINITY; l_run[qt][r] = 0.f; }
  }

  for (int j0 = 0; j0 < SEQ; j0 += 32) {
    __syncthreads();
    // ---- Stage K rows j0..j0+31 (contiguous 4KB) via async DMA to LDS. ----
#if USE_ASYNC_LDS
    {
      // Generic LDS address carries the byte offset in bits [31:0].
      unsigned loff = (unsigned)(size_t)&Ksh[t * 8];
      unsigned long long gaddr =
          (unsigned long long)(size_t)&K[(size_t)j0 * D_PAD + t * 8];
      asm volatile("global_load_async_to_lds_b128 %0, %1, off"
                   :: "v"(loff), "v"(gaddr)
                   : "memory");
    }
#else
    *(uint4*)&Ksh[t * 8] = *(const uint4*)&K[(size_t)j0 * D_PAD + t * 8];
#endif
    // ---- Stage V transposed: Vtsh[feature][key]. ----
    {
      int key = t >> 3, f0 = (t & 7) * 8;
      v8h vv = *(const v8h*)&V[(size_t)(j0 + key) * D_PAD + f0];
#pragma unroll
      for (int i = 0; i < 8; ++i) Vtsh[(f0 + i) * 32 + key] = vv[i];
    }
    if (j0 + 32 < SEQ) {
      __builtin_prefetch(&K[(size_t)(j0 + 32) * D_PAD + t * 8], 0, 0);
      __builtin_prefetch(&V[(size_t)(j0 + 32) * D_PAD + t * 8], 0, 0);
    }
#if USE_ASYNC_LDS
    asm volatile("s_wait_asynccnt 0x0" ::: "memory");
#endif
    __syncthreads();

#pragma unroll
    for (int qt = 0; qt < QTILES; ++qt) {
      // S = Q @ Kblock^T : 16x32 logits as two 16x16 accumulators.
      v8f s0 = zero, s1 = zero;
#pragma unroll
      for (int kc = 0; kc < 2; ++kc) {
        int kb = kc * 32 + hi * 8;
        const _Float16* k0p = &Ksh[(size_t)lm * D_PAD + kb];
        const _Float16* k1p = &Ksh[(size_t)(16 + lm) * D_PAD + kb];
        v16h b0 = cat8(*(const v8h*)k0p, *(const v8h*)(k0p + 16));
        v16h b1 = cat8(*(const v8h*)k1p, *(const v8h*)(k1p + 16));
        s0 = wmma_f16(qa[qt][kc], b0, s0);
        s1 = wmma_f16(qa[qt][kc], b1, s1);
      }
      // Online softmax: row r+8*hi lives in vgpr r across this lane half.
#pragma unroll
      for (int r = 0; r < 8; ++r) {
        float a0 = s0[r], a1 = s1[r];
        float mx = fmaxf(a0, a1);
#pragma unroll
        for (int o = 8; o >= 1; o >>= 1) mx = fmaxf(mx, __shfl_xor(mx, o, 16));
        float mnew = fmaxf(m_run[qt][r], mx);
        float alpha = __expf(m_run[qt][r] - mnew);  // first block: exp(-inf)=0
        m_run[qt][r] = mnew;
        float p0 = __expf(a0 - mnew), p1 = __expf(a1 - mnew);
        float ls = p0 + p1;
#pragma unroll
        for (int o = 8; o >= 1; o >>= 1) ls += __shfl_xor(ls, o, 16);
        l_run[qt][r] = l_run[qt][r] * alpha + ls;
#pragma unroll
        for (int ft = 0; ft < 4; ++ft) acc[qt][ft][r] *= alpha;
        int row = r + hi * 8;
        Psh[wave][qt][row * 32 + lm]      = (_Float16)p0;
        Psh[wave][qt][row * 32 + 16 + lm] = (_Float16)p1;
      }
    }

    // P (C-layout) -> A-layout via wave-private LDS, then acc += P @ V.
    v16h pa[QTILES];
#pragma unroll
    for (int qt = 0; qt < QTILES; ++qt) {
      const _Float16* base = &Psh[wave][qt][lm * 32 + hi * 8];
      pa[qt] = cat8(*(const v8h*)base, *(const v8h*)(base + 16));
    }
#pragma unroll
    for (int ft = 0; ft < 4; ++ft) {
      const _Float16* vp = &Vtsh[(ft * 16 + lm) * 32 + hi * 8];
      v16h vb = cat8(*(const v8h*)vp, *(const v8h*)(vp + 16));
#pragma unroll
      for (int qt = 0; qt < QTILES; ++qt)
        acc[qt][ft] = wmma_f16(pa[qt], vb, acc[qt][ft]);
    }
  }

  // Epilogue: normalize and write [b][s][h*40+c] f32.
#pragma unroll
  for (int qt = 0; qt < QTILES; ++qt)
#pragma unroll
    for (int r = 0; r < 8; ++r) {
      float inv = 1.f / l_run[qt][r];
      int srow = q0 + qt * 16 + r + hi * 8;
#pragma unroll
      for (int ft = 0; ft < 4; ++ft) {
        int cc = ft * 16 + lm;
        if (cc < D_HEAD)
          oat[((size_t)b * SEQ + srow) * D_MODEL + h * D_HEAD + cc] =
              acc[qt][ft][r] * inv;
      }
    }
}

// ---------------------------------------------------------------------------
// Workspace: [0,4M) q_f16 [8][4096][64] | [4M,8M) k_f16 | [8M,20M) v_f16
// [24][4096][64] | [20M,35M) attn-out f32 [3][4096][320].
// ---------------------------------------------------------------------------
extern "C" void kernel_launch(void* const* d_in, const int* in_sizes, int n_in,
                              void* d_out, int out_size, void* d_ws,
                              size_t ws_size, hipStream_t stream) {
  (void)in_sizes; (void)n_in; (void)out_size; (void)ws_size;
  const float* x  = (const float*)d_in[0];
  const float* Wq = (const float*)d_in[1];
  const float* Wk = (const float*)d_in[2];
  const float* Wv = (const float*)d_in[3];
  const float* Wo = (const float*)d_in[4];
  const float* bo = (const float*)d_in[5];

  const size_t plane = (size_t)SEQ * D_PAD;
  _Float16* qws = (_Float16*)d_ws;
  _Float16* kws = qws + (size_t)N_HEADS * plane;
  _Float16* vws = kws + (size_t)N_HEADS * plane;
  const size_t f16_elems = (size_t)(2 * N_HEADS + BATCH * N_HEADS) * plane;
  float* oat = (float*)((char*)d_ws + f16_elems * sizeof(_Float16));

  hipMemsetAsync(d_ws, 0, f16_elems * sizeof(_Float16), stream);

  const int tilesQK = (SEQ / 16) * (D_MODEL / 16) / 8;          // 640
  const int tilesV  = (BATCH * SEQ / 16) * (D_MODEL / 16) / 8;  // 1920
  proj_heads_f16<<<tilesQK, 256, 0, stream>>>(x, Wq, qws, QK_SCALE);
  proj_heads_f16<<<tilesQK, 256, 0, stream>>>(x, Wk, kws, 1.0f);
  proj_heads_f16<<<tilesV, 256, 0, stream>>>(x, Wv, vws, 1.0f);

  flash_attn<<<dim3(SEQ / (128 * QTILES), BATCH * N_HEADS), 256, 0, stream>>>(
      qws, kws, vws, oat);

  proj_out_f32<<<tilesV, 256, 0, stream>>>(oat, Wo, bo, (float*)d_out);
}
